// EPIModel_39402029973926
// MI455X (gfx1250) — matrix operand order repaired and checked by reading, not compiled, not run on hardware
//
#include <hip/hip_runtime.h>
#include <math.h>

// ---------------- model constants ----------------
#define VV    4097
#define DD    128
#define OUTD  64
#define HH    8
#define FFD   256
#define LL    2
#define KK    9
#define POOL  20
#define BB    32
#define LE_   3000
#define LP_   2000
#define S_EN  149          // (3000-9+1 - 20)/20 + 1
#define S_PR  99           // (2000-9+1 - 20)/20 + 1
#define GRIDF 5
#define WIDTHF 257
#define DRD   8
#define CONVK (DD*KK)      // 1152

typedef __attribute__((ext_vector_type(16))) _Float16 v16h;
typedef __attribute__((ext_vector_type(8)))  float    v8f;

#define DEV static __device__ __forceinline__

// ---------------- WMMA fragment helpers (CDNA5 16x16x32 f16 layouts) -------
// A fragment: 16x32 f16.  lanes 0-15: M=lane, halves 0-7 = K 0-7, halves 8-15 = K 16-23
//             lanes 16-31: M=lane-16, halves 0-7 = K 8-15, halves 8-15 = K 24-31
// (all GEMM M are multiples of 16 here: B=32 -> no row clamping needed)
DEV v16h load_a_f32(const float* A, int lda, int row0, int k0) {
  const int lane = threadIdx.x & 31;
  const int m = row0 + (lane & 15);
  const float* ap = A + (size_t)m * lda + k0 + ((lane >> 4) << 3);
  v16h a;
#pragma unroll
  for (int e = 0; e < 8; ++e) a[e] = (_Float16)ap[e];
#pragma unroll
  for (int e = 0; e < 8; ++e) a[8 + e] = (_Float16)ap[16 + e];
  return a;
}

// B fragment from pre-packed weights: Bp[((k>>5)*N + n)*32 + (k&31)].
// lane needs K = k0 + (lane>>4)*8 + e (e<8) and K = k0 + 16 + (lane>>4)*8 + (e-8) (e>=8).
DEV v16h load_b_pk(const _Float16* Bp, int N, int k0, int col0) {
  const int lane = threadIdx.x & 31;
  const int n = col0 + (lane & 15);
  const _Float16* bp = Bp + (((size_t)(k0 >> 5)) * N + n) * 32 + ((lane >> 4) << 3);
  v16h b;
#pragma unroll
  for (int e = 0; e < 8; ++e) { b[e] = bp[e]; b[8 + e] = bp[16 + e]; }
  return b;
}

// D tile: 16x16 f32, 8 VGPRs: vgpr r -> M = r + 8*(lane>=16), N = lane&15
DEV void store_d(float* C, int ldc, int row0, int col0, v8f acc,
                 const float* bias, int act) {
  const int lane = threadIdx.x & 31;
  const int n = col0 + (lane & 15);
  const int rbase = row0 + ((lane >> 4) << 3);
  const float bz = bias ? bias[n] : 0.f;
#pragma unroll
  for (int r = 0; r < 8; ++r) {
    float v = acc[r] + bz;
    if (act == 1) v = fmaxf(v, 0.f);
    C[(size_t)(rbase + r) * ldc + n] = v;
  }
}

// ---------------- weight pre-packing ----------------
// mode 0: src is K x N row-major (x @ W)
// mode 1: src is N x K row-major (x @ W.T)
// mode 2: conv weights W[o, c, k] with GEMM K index kk = k*128 + c
__global__ void k_pack(const float* __restrict__ src, _Float16* __restrict__ dst,
                       int Kd, int N, int mode) {
  const int idx = blockIdx.x * blockDim.x + threadIdx.x;
  if (idx >= Kd * N) return;
  const int k = idx / N, n = idx - k * N;
  float v;
  if (mode == 0)      v = src[(size_t)k * N + n];
  else if (mode == 1) v = src[(size_t)n * Kd + k];
  else {
    const int c = k & 127, kp = k >> 7;
    v = src[(size_t)n * CONVK + c * KK + kp];
  }
  dst[(((size_t)(k >> 5)) * N + n) * 32 + (k & 31)] = (_Float16)v;
}

__global__ void k_cvt_f16(const float* __restrict__ src, _Float16* __restrict__ dst, int n) {
  const int i = blockIdx.x * blockDim.x + threadIdx.x;
  if (i < n) dst[i] = (_Float16)src[i];
}

// ---------------- generic WMMA GEMM (packed f16 weights) ----------------
__global__ void k_gemm_pk(const float* __restrict__ A, const _Float16* __restrict__ Bp,
                          const float* __restrict__ bias, float* __restrict__ C,
                          int M, int N, int Kd, int act) {
  const int wid = (int)((blockIdx.x * blockDim.x + threadIdx.x) >> 5);
  const int ntn = N >> 4;
  const int mt = wid / ntn;
  const int row0 = mt << 4;
  if (row0 >= M) return;                      // wave-uniform
  const int col0 = (wid - mt * ntn) << 4;
  v8f acc = {0.f,0.f,0.f,0.f,0.f,0.f,0.f,0.f};
  for (int k0 = 0; k0 < Kd; k0 += 32) {
    if (k0 + 32 < Kd)
      __builtin_prefetch(A + (size_t)row0 * Kd + k0 + 32, 0, 0);   // global_prefetch_b8
    v16h a = load_a_f32(A, Kd, row0, k0);
    v16h b = load_b_pk(Bp, N, k0, col0);
    acc = __builtin_amdgcn_wmma_f32_16x16x32_f16(false, a, false, b,
                                                 (short)0, acc, false, false);
  }
  store_d(C, N, row0, col0, acc, bias, act);
}

// ---------------- frontend: implicit-im2col conv GEMM (f16 emb gather) -----
// conv[b,t,o] = relu( sum_{c,k} emb[ids[b,t+k],c] * W[o,c,k] + bias[o] )
__global__ void k_conv_gemm(const int* __restrict__ ids, const _Float16* __restrict__ emb16,
                            const _Float16* __restrict__ Wp, const float* __restrict__ bias,
                            float* __restrict__ C, int Mtot, int Lc2, int Lids) {
  const int wid = (int)((blockIdx.x * blockDim.x + threadIdx.x) >> 5);
  const int mt = wid >> 2;
  const int row0 = mt << 4;
  if (row0 >= Mtot) return;                   // wave-uniform; Mtot % 16 == 0
  const int col0 = (wid & 3) << 4;
  const int lane = threadIdx.x & 31;
  const int m = row0 + (lane & 15);
  const int bb = m / Lc2;
  const int tt = m - bb * Lc2;
  const int* idrow = ids + (size_t)bb * Lids + tt;
  const int kbA = (lane >> 4) << 3;
  v8f acc = {0.f,0.f,0.f,0.f,0.f,0.f,0.f,0.f};
  for (int k0 = 0; k0 < CONVK; k0 += 32) {
    v16h a;
    {   // gathered A fragment: two 8-half contiguous runs (c contiguous, same id row)
      const int kk0 = k0 + kbA;               // halves 0-7:  K = kk0 .. kk0+7
      const int kk1 = kk0 + 16;               // halves 8-15: K = kk1 .. kk1+7
      const _Float16* p0 = emb16 + (size_t)idrow[kk0 >> 7] * DD + (kk0 & 127);
      const _Float16* p1 = emb16 + (size_t)idrow[kk1 >> 7] * DD + (kk1 & 127);
#pragma unroll
      for (int e = 0; e < 8; ++e) a[e] = p0[e];
#pragma unroll
      for (int e = 0; e < 8; ++e) a[8 + e] = p1[e];
    }
    v16h b = load_b_pk(Wp, OUTD, k0, col0);
    acc = __builtin_amdgcn_wmma_f32_16x16x32_f16(false, a, false, b,
                                                 (short)0, acc, false, false);
  }
  store_d(C, OUTD, row0, col0, acc, bias, 1 /*relu*/);
}

// maxpool(20) + batchnorm
__global__ void k_pool_bn(const float* __restrict__ conv, const float* __restrict__ bn,
                          float* __restrict__ x, int Ntok) {
  const int t = blockIdx.x * blockDim.x + threadIdx.x;
  if (t >= Ntok * OUTD) return;
  const int o = t & 63;
  const int tok = t >> 6;
  const float* cp = conv + (size_t)tok * POOL * OUTD + o;
  float mx = -3.0e38f;
  for (int j = 0; j < POOL; ++j) mx = fmaxf(mx, cp[j * OUTD]);
  const float g = bn[o], be = bn[OUTD + o], mu = bn[2 * OUTD + o], var = bn[3 * OUTD + o];
  x[t] = (mx - mu) * rsqrtf(var + 1e-5f) * g + be;
}

// ---------------- RoPE (hd=8 per head) ----------------
__global__ void k_rope(float* __restrict__ q, float* __restrict__ k, int Ntok, int S) {
  const int idx = blockIdx.x * blockDim.x + threadIdx.x;    // tok*32 + h*4 + i
  if (idx >= Ntok * 32) return;
  const int i = idx & 3;
  const int h = (idx >> 2) & 7;
  const int tok = idx >> 5;
  const int s = tok % S;
  const float inv = __powf(10000.f, -(2.f * (float)i) / 8.f);
  const float ang = (float)s * inv;
  const float c = __cosf(ang), sn = __sinf(ang);
  float* qp = q + (size_t)tok * OUTD + h * 8 + 2 * i;
  float x1 = qp[0], x2 = qp[1];
  qp[0] = x1 * c - x2 * sn; qp[1] = x1 * sn + x2 * c;
  float* kp = k + (size_t)tok * OUTD + h * 8 + 2 * i;
  x1 = kp[0]; x2 = kp[1];
  kp[0] = x1 * c - x2 * sn; kp[1] = x1 * sn + x2 * c;
}

// ---------------- attention (hd = 8, scalar; 2-pass softmax) ----------------
__global__ void k_attn(const float* __restrict__ q, const float* __restrict__ kmat,
                       const float* __restrict__ v, float* __restrict__ ctx,
                       int Sq, int Sk, float* aloss, float alossScale) {
  const int b = blockIdx.x >> 3;
  const int h = blockIdx.x & 7;
  const int i = threadIdx.x;
  if (i >= Sq) return;
  const float scale = 0.35355339059327373f;   // 1/sqrt(8)
  const float* qp = q + (size_t)(b * Sq + i) * OUTD + h * 8;
  const float* kbase = kmat + (size_t)b * Sk * OUTD + h * 8;
  const float* vbase = v + (size_t)b * Sk * OUTD + h * 8;
  float qr[8];
#pragma unroll
  for (int d = 0; d < 8; ++d) qr[d] = qp[d];
  float mx = -3.0e38f;
  for (int j = 0; j < Sk; ++j) {
    const float* kp = kbase + (size_t)j * OUTD;
    float dot = 0.f;
#pragma unroll
    for (int d = 0; d < 8; ++d) dot += qr[d] * kp[d];
    mx = fmaxf(mx, dot * scale);
  }
  float sum = 0.f, sumsq = 0.f;
  float acc[8] = {0.f,0.f,0.f,0.f,0.f,0.f,0.f,0.f};
  for (int j = 0; j < Sk; ++j) {
    const float* kp = kbase + (size_t)j * OUTD;
    float dot = 0.f;
#pragma unroll
    for (int d = 0; d < 8; ++d) dot += qr[d] * kp[d];
    const float p = __expf(dot * scale - mx);
    sum += p; sumsq += p * p;
    const float* vp = vbase + (size_t)j * OUTD;
#pragma unroll
    for (int d = 0; d < 8; ++d) acc[d] += p * vp[d];
  }
  const float inv = 1.f / sum;
  float* cp = ctx + (size_t)(b * Sq + i) * OUTD + h * 8;
#pragma unroll
  for (int d = 0; d < 8; ++d) cp[d] = acc[d] * inv;
  if (aloss) atomicAdd(aloss, sumsq * inv * inv * alossScale);
}

// ---------------- channel attention gate ----------------
__global__ void k_ca(const float* __restrict__ ctx, const float* __restrict__ w1,
                     const float* __restrict__ w2, float* __restrict__ ca, int S) {
  __shared__ float avg[OUTD], mxs[OUTD], ha[DRD], hm[DRD];
  const int b = blockIdx.x, o = threadIdx.x;   // 64 threads
  float a = 0.f, m = -3.0e38f;
  const float* base = ctx + (size_t)b * S * OUTD + o;
  for (int s = 0; s < S; ++s) { const float vv = base[(size_t)s * OUTD]; a += vv; m = fmaxf(m, vv); }
  avg[o] = a / (float)S; mxs[o] = m;
  __syncthreads();
  if (o < DRD) {
    float sa = 0.f, sm = 0.f;
    for (int c = 0; c < OUTD; ++c) { sa += avg[c] * w1[c * DRD + o]; sm += mxs[c] * w1[c * DRD + o]; }
    ha[o] = fmaxf(sa, 0.f); hm[o] = fmaxf(sm, 0.f);
  }
  __syncthreads();
  float z = 0.f;
  for (int r = 0; r < DRD; ++r) z += (ha[r] + hm[r]) * w2[r * OUTD + o];
  ca[b * OUTD + o] = 1.f / (1.f + __expf(-z));
}

// apply ca, produce per-token mean/max features
__global__ void k_gate1(float* __restrict__ ctx, const float* __restrict__ ca,
                        float* __restrict__ fmean, float* __restrict__ fmax, int S) {
  __shared__ float r1[OUTD], r2[OUTD];
  const int tok = blockIdx.x;
  const int b = tok / S;
  const int o = threadIdx.x;
  float vv = ctx[(size_t)tok * OUTD + o] * ca[b * OUTD + o];
  ctx[(size_t)tok * OUTD + o] = vv;
  r1[o] = vv; r2[o] = vv;
  __syncthreads();
  for (int st = 32; st > 0; st >>= 1) {
    if (o < st) { r1[o] += r1[o + st]; r2[o] = fmaxf(r2[o], r2[o + st]); }
    __syncthreads();
  }
  if (o == 0) { fmean[tok] = r1[0] / 64.f; fmax[tok] = r2[0]; }
}

// spatial gate: conv7 'SAME' over the 2-channel feature, sigmoid, scale ctx
__global__ void k_gate2(float* __restrict__ ctx, const float* __restrict__ fmean,
                        const float* __restrict__ fmax, const float* __restrict__ saw,
                        int Ntok, int S) {
  const int tok = blockIdx.x * blockDim.x + threadIdx.x;
  if (tok >= Ntok) return;
  const int b = tok / S, s = tok - b * S;
  float z = 0.f;
#pragma unroll
  for (int t = -3; t <= 3; ++t) {
    const int ss = s + t;
    if (ss >= 0 && ss < S)
      z += saw[t + 3] * fmean[b * S + ss] + saw[7 + t + 3] * fmax[b * S + ss];
  }
  const float sa = 1.f / (1.f + __expf(-z));
  float* cp = ctx + (size_t)tok * OUTD;
  for (int o = 0; o < OUTD; ++o) cp[o] *= sa;
}

// x = LayerNorm(x + y) * gamma + beta   (over 64 channels)
__global__ void k_add_ln(float* __restrict__ x, const float* __restrict__ y,
                         const float* __restrict__ gamma, const float* __restrict__ beta) {
  __shared__ float red[OUTD];
  const int tok = blockIdx.x, o = threadIdx.x;
  const float v = x[(size_t)tok * OUTD + o] + y[(size_t)tok * OUTD + o];
  red[o] = v;
  __syncthreads();
  for (int st = 32; st > 0; st >>= 1) { if (o < st) red[o] += red[o + st]; __syncthreads(); }
  const float mu = red[0] / 64.f;
  __syncthreads();
  const float dv = v - mu;
  red[o] = dv * dv;
  __syncthreads();
  for (int st = 32; st > 0; st >>= 1) { if (o < st) red[o] += red[o + st]; __syncthreads(); }
  const float var = red[0] / 64.f;
  x[(size_t)tok * OUTD + o] = dv * rsqrtf(var + 1e-5f) * gamma[o] + beta[o];
}

// mean over sequence -> combined slice (row stride 128)
__global__ void k_seqmean(const float* __restrict__ xin, float* __restrict__ dst, int S) {
  const int b = blockIdx.x, o = threadIdx.x;   // 64 threads
  float a = 0.f;
  const float* base = xin + (size_t)b * S * OUTD + o;
  for (int s = 0; s < S; ++s) a += base[(size_t)s * OUTD];
  dst[b * 128 + o] = a / (float)S;
}

// Fourier feature layer
__global__ void k_fourier(const float* __restrict__ xin, const float* __restrict__ wc,
                          const float* __restrict__ ws, const float* __restrict__ bias,
                          float* __restrict__ outp, int In, int On, int applySig) {
  const int idx = blockIdx.x * blockDim.x + threadIdx.x;
  if (idx >= BB * On) return;
  const int b = idx / On, o = idx - b * On;
  float acc = bias[o];
  const float* xb = xin + (size_t)b * In;
  for (int i = 0; i < In; ++i) {
    const float xv = xb[i];
    const float* wcp = wc + ((size_t)o * In + i) * GRIDF;
    const float* wsp = ws + ((size_t)o * In + i) * GRIDF;
#pragma unroll
    for (int g = 0; g < GRIDF; ++g) {
      const float kx = xv * (float)(g + 1);
      acc += __cosf(kx) * wcp[g] + __sinf(kx) * wsp[g];
    }
  }
  if (applySig) acc = 1.f / (1.f + __expf(-acc));
  outp[b * On + o] = acc;
}

__global__ void k_finalize(const float* __restrict__ aloss, float* __restrict__ outp) {
  outp[BB + BB * 128] = aloss[0];   // d_out[4128]
}

// ---------------- host orchestration ----------------
extern "C" void kernel_launch(void* const* d_in, const int* in_sizes, int n_in,
                              void* d_out, int out_size, void* d_ws, size_t ws_size,
                              hipStream_t stream) {
  (void)in_sizes; (void)n_in; (void)out_size; (void)ws_size;
  const int*   ids_en   = (const int*)  d_in[0];
  const int*   ids_pr   = (const int*)  d_in[1];
  const float* emb_en   = (const float*)d_in[4];
  const float* emb_pr   = (const float*)d_in[5];
  const float* conv_w_en= (const float*)d_in[6];
  const float* conv_b_en= (const float*)d_in[7];
  const float* conv_w_pr= (const float*)d_in[8];
  const float* conv_b_pr= (const float*)d_in[9];
  const float* bn_en    = (const float*)d_in[10];
  const float* bn_pr    = (const float*)d_in[11];
  const float* t_qkv_w  = (const float*)d_in[12];
  const float* t_qkv_b  = (const float*)d_in[13];
  const float* t_o_w    = (const float*)d_in[14];
  const float* t_o_b    = (const float*)d_in[15];
  const float* t_ca_w1  = (const float*)d_in[16];
  const float* t_ca_w2  = (const float*)d_in[17];
  const float* t_sa_w   = (const float*)d_in[18];
  const float* t_ff1_w  = (const float*)d_in[19];
  const float* t_ff1_b  = (const float*)d_in[20];
  const float* t_ff2_w  = (const float*)d_in[21];
  const float* t_ff2_b  = (const float*)d_in[22];
  const float* t_ln1    = (const float*)d_in[23];
  const float* t_ln2    = (const float*)d_in[24];
  const float* cross_in_w = (const float*)d_in[25];
  const float* cross_in_b = (const float*)d_in[26];
  const float* cross_out_w= (const float*)d_in[27];
  const float* cross_out_b= (const float*)d_in[28];
  const float* self_in_w  = (const float*)d_in[29];
  const float* self_in_b  = (const float*)d_in[30];
  const float* self_out_w = (const float*)d_in[31];
  const float* self_out_b = (const float*)d_in[32];
  const float* fk1_c = (const float*)d_in[33];
  const float* fk1_s = (const float*)d_in[34];
  const float* fk1_b = (const float*)d_in[35];
  const float* fk2_c = (const float*)d_in[36];
  const float* fk2_s = (const float*)d_in[37];
  const float* fk2_b = (const float*)d_in[38];

  float* out = (float*)d_out;
  float* combined = out + BB;                  // d_out[32 .. 32+4095]

  // ---- workspace arena (f32 region, then f16 region) ----
  float* W = (float*)d_ws;
  size_t off = 0;
  auto alloc = [&](size_t n) { float* p = W + off; off += n; return p; };
  float* x_en   = alloc((size_t)BB * S_EN * OUTD);
  float* x_pr   = alloc((size_t)BB * S_PR * OUTD);
  float* en_att = alloc((size_t)BB * S_EN * OUTD);
  float* ca     = alloc((size_t)BB * OUTD);
  float* fmean  = alloc((size_t)BB * S_EN);
  float* fmaxb  = alloc((size_t)BB * S_EN);
  float* hidden = alloc((size_t)BB * WIDTHF);
  float* aloss  = alloc(1);
  float* scratch = alloc((size_t)BB * S_EN * POOL * OUTD);   // conv acts / layer temps
  float* q   = scratch;
  float* kqt = scratch + 1 * (size_t)BB * S_EN * OUTD;
  float* v   = scratch + 2 * (size_t)BB * S_EN * OUTD;
  float* ctx = scratch + 3 * (size_t)BB * S_EN * OUTD;
  float* tmp = scratch + 4 * (size_t)BB * S_EN * OUTD;
  float* h1  = scratch + 5 * (size_t)BB * S_EN * OUTD;       // B*S*256 FF hidden

  // f16 pool (packed weights + f16 embeddings)
  _Float16* HP = (_Float16*)(W + off);
  size_t hoff = 0;
  auto halloc = [&](size_t n) { _Float16* p = HP + hoff; hoff += n; return p; };
  _Float16* emb16_en = halloc((size_t)VV * DD);
  _Float16* emb16_pr = halloc((size_t)VV * DD);
  _Float16* convp_en = halloc((size_t)(CONVK / 32) * OUTD * 32);   // 73728
  _Float16* convp_pr = halloc((size_t)(CONVK / 32) * OUTD * 32);
  _Float16* qkvp = halloc((size_t)12 * 4096);   // (s,l,i): Kd=64,N=64 -> 4096 each
  _Float16* owp  = halloc((size_t)4 * 4096);
  _Float16* ff1p = halloc((size_t)4 * 16384);   // Kd=64,N=256
  _Float16* ff2p = halloc((size_t)4 * 16384);   // Kd=256,N=64
  _Float16* cinp = halloc((size_t)3 * 4096);
  _Float16* coutp= halloc((size_t)1 * 4096);
  _Float16* sinp = halloc((size_t)3 * 4096);
  _Float16* soutp= halloc((size_t)1 * 4096);

  auto pack = [&](const float* src, _Float16* dst, int Kd, int N, int mode) {
    k_pack<<<(Kd * N + 255) / 256, 256, 0, stream>>>(src, dst, Kd, N, mode);
  };
  auto gemm = [&](const float* A, const _Float16* Bp, const float* bias, float* C,
                  int M, int N, int Kd, int act) {
    const int waves = (M / 16) * (N / 16);
    k_gemm_pk<<<(waves + 7) / 8, 256, 0, stream>>>(A, Bp, bias, C, M, N, Kd, act);
  };

  // ---------------- weight pre-pack / embedding cvt ----------------
  k_cvt_f16<<<((VV * DD) + 255) / 256, 256, 0, stream>>>(emb_en, emb16_en, VV * DD);
  k_cvt_f16<<<((VV * DD) + 255) / 256, 256, 0, stream>>>(emb_pr, emb16_pr, VV * DD);
  pack(conv_w_en, convp_en, CONVK, OUTD, 2);
  pack(conv_w_pr, convp_pr, CONVK, OUTD, 2);
  for (int sl = 0; sl < 4; ++sl) {
    for (int i = 0; i < 3; ++i)
      pack(t_qkv_w + ((size_t)sl * 3 + i) * 4096, qkvp + ((size_t)sl * 3 + i) * 4096, 64, 64, 0);
    pack(t_o_w  + (size_t)sl * 4096,      owp  + (size_t)sl * 4096,  64, 64, 0);
    pack(t_ff1_w + (size_t)sl * 64 * 256, ff1p + (size_t)sl * 16384, 64, 256, 0);
    pack(t_ff2_w + (size_t)sl * 256 * 64, ff2p + (size_t)sl * 16384, 256, 64, 0);
  }
  for (int i = 0; i < 3; ++i) {
    pack(cross_in_w + (size_t)i * 4096, cinp + (size_t)i * 4096, 64, 64, 1);
    pack(self_in_w  + (size_t)i * 4096, sinp + (size_t)i * 4096, 64, 64, 1);
  }
  pack(cross_out_w, coutp, 64, 64, 1);
  pack(self_out_w,  soutp, 64, 64, 1);

  // ---------------- frontend ----------------
  {
    const int Mc = BB * S_EN * POOL;                 // 95360 (mult of 16)
    const int waves = (Mc / 16) * 4;
    k_conv_gemm<<<(waves + 7) / 8, 256, 0, stream>>>(ids_en, emb16_en, convp_en, conv_b_en,
                                                     scratch, Mc, S_EN * POOL, LE_);
    const int nt = BB * S_EN;
    k_pool_bn<<<(nt * OUTD + 255) / 256, 256, 0, stream>>>(scratch, bn_en, x_en, nt);
  }
  {
    const int Mc = BB * S_PR * POOL;                 // 63360 (mult of 16)
    const int waves = (Mc / 16) * 4;
    k_conv_gemm<<<(waves + 7) / 8, 256, 0, stream>>>(ids_pr, emb16_pr, convp_pr, conv_b_pr,
                                                     scratch, Mc, S_PR * POOL, LP_);
    const int nt = BB * S_PR;
    k_pool_bn<<<(nt * OUTD + 255) / 256, 256, 0, stream>>>(scratch, bn_pr, x_pr, nt);
  }
  hipMemsetAsync(aloss, 0, sizeof(float), stream);

  // ---------------- CBAT transformer layers ----------------
  auto cbat = [&](float* x, int S, int sidx, int l) {
    const int M = BB * S;
    const size_t sl = (size_t)sidx * LL + l;
    const float* qkvb = t_qkv_b + sl * 3 * 64;
    gemm(x, qkvp + (sl * 3 + 0) * 4096, qkvb,       q,   M, 64, 64, 0);
    gemm(x, qkvp + (sl * 3 + 1) * 4096, qkvb + 64,  kqt, M, 64, 64, 0);
    gemm(x, qkvp + (sl * 3 + 2) * 4096, qkvb + 128, v,   M, 64, 64, 0);
    k_rope<<<(M * 32 + 255) / 256, 256, 0, stream>>>(q, kqt, M, S);
    const int th = ((S + 31) / 32) * 32;
    k_attn<<<BB * HH, th, 0, stream>>>(q, kqt, v, ctx, S, S, aloss,
                                       1.f / ((float)BB * HH * S * S));
    k_ca<<<BB, OUTD, 0, stream>>>(ctx, t_ca_w1 + sl * 64 * 8, t_ca_w2 + sl * 8 * 64, ca, S);
    k_gate1<<<M, OUTD, 0, stream>>>(ctx, ca, fmean, fmaxb, S);
    k_gate2<<<(M + 255) / 256, 256, 0, stream>>>(ctx, fmean, fmaxb, t_sa_w + sl * 14, M, S);
    gemm(ctx, owp + sl * 4096, t_o_b + sl * 64, tmp, M, 64, 64, 0);
    k_add_ln<<<M, OUTD, 0, stream>>>(x, tmp, t_ln1 + sl * 2 * 64, t_ln1 + sl * 2 * 64 + 64);
    gemm(x, ff1p + sl * 16384, t_ff1_b + sl * 256, h1, M, 256, 64, 1);
    gemm(h1, ff2p + sl * 16384, t_ff2_b + sl * 64, tmp, M, 64, 256, 0);
    k_add_ln<<<M, OUTD, 0, stream>>>(x, tmp, t_ln2 + sl * 2 * 64, t_ln2 + sl * 2 * 64 + 64);
  };
  for (int l = 0; l < LL; ++l) cbat(x_en, S_EN, 0, l);
  for (int l = 0; l < LL; ++l) cbat(x_pr, S_PR, 1, l);

  // ---------------- cross / self MHA ----------------
  auto mha = [&](const float* qin, int Sq, const float* kin, int Sk,
                 const _Float16* inp, const float* inb, const _Float16* outp_,
                 const float* outb, float* dst) {
    const int Mq = BB * Sq, Mk = BB * Sk;
    gemm(qin, inp,            inb,       q,   Mq, 64, 64, 0);   // Wq.T
    gemm(kin, inp + 4096,     inb + 64,  kqt, Mk, 64, 64, 0);   // Wk.T
    gemm(kin, inp + 2 * 4096, inb + 128, v,   Mk, 64, 64, 0);   // Wv.T
    const int th = ((Sq + 31) / 32) * 32;
    k_attn<<<BB * HH, th, 0, stream>>>(q, kqt, v, ctx, Sq, Sk, nullptr, 0.f);
    gemm(ctx, outp_, outb, dst, Mq, 64, 64, 0);                 // out_w.T
  };
  mha(x_en, S_EN, x_pr, S_PR, cinp, cross_in_b, coutp, cross_out_b, en_att);
  mha(en_att, S_EN, en_att, S_EN, sinp, self_in_b, soutp, self_out_b, tmp);
  k_seqmean<<<BB, OUTD, 0, stream>>>(tmp, combined, S_EN);
  mha(x_pr, S_PR, x_pr, S_PR, sinp, self_in_b, soutp, self_out_b, tmp);
  k_seqmean<<<BB, OUTD, 0, stream>>>(tmp, combined + OUTD, S_PR);

  // ---------------- Fourier head ----------------
  k_fourier<<<(BB * WIDTHF + 255) / 256, 256, 0, stream>>>(combined, fk1_c, fk1_s, fk1_b,
                                                           hidden, 2 * OUTD, WIDTHF, 0);
  k_fourier<<<1, BB, 0, stream>>>(hidden, fk2_c, fk2_s, fk2_b, out, WIDTHF, 1, 1);
  k_finalize<<<1, 1, 0, stream>>>(aloss, out);
}
